// LongTermMemory_6279242187107
// MI455X (gfx1250) — matrix operand order repaired and checked by reading, not compiled, not run on hardware
//
#include <hip/hip_runtime.h>
#include <hip/hip_bf16.h>

// ---------------------------------------------------------------------------
// Types for CDNA5 WMMA
// ---------------------------------------------------------------------------
typedef __attribute__((ext_vector_type(16))) __bf16 v16bf;
typedef __attribute__((ext_vector_type(8)))  float  v8f;

static __device__ __forceinline__ unsigned short f32_to_bf16_bits(float f) {
    unsigned u = __builtin_bit_cast(unsigned, f);
    u += 0x7FFFu + ((u >> 16) & 1u);   // round-to-nearest-even
    return (unsigned short)(u >> 16);
}
static __device__ __forceinline__ __bf16 f2bf(float f) {
    unsigned short s = f32_to_bf16_bits(f);
    return __builtin_bit_cast(__bf16, s);
}

// CDNA5 async global->LDS copy (ASYNCcnt-tracked). Each active lane copies
// 16 bytes from its global address to its LDS address.
static __device__ __forceinline__ void async_copy_b128(unsigned lds_addr,
                                                       const __bf16* gptr) {
    asm volatile("global_load_async_to_lds_b128 %0, %1, off"
                 :
                 : "v"(lds_addr), "v"(gptr)
                 : "memory");
}
#define WAIT_ASYNCCNT(n) asm volatile("s_wait_asynccnt " #n ::: "memory")

// ---------------------------------------------------------------------------
// fp32 -> bf16 conversion, 4 elements per thread (x / activations path)
// ---------------------------------------------------------------------------
__global__ __launch_bounds__(256) void cvt_f32_bf16_x4(const float* __restrict__ in,
                                                       __bf16* __restrict__ out,
                                                       int n4) {
    int i = blockIdx.x * 256 + threadIdx.x;
    if (i < n4) {
        float4 f = reinterpret_cast<const float4*>(in)[i];
        ushort4 u;
        u.x = f32_to_bf16_bits(f.x);
        u.y = f32_to_bf16_bits(f.y);
        u.z = f32_to_bf16_bits(f.z);
        u.w = f32_to_bf16_bits(f.w);
        reinterpret_cast<ushort4*>(out)[i] = u;
    }
}

// ---------------------------------------------------------------------------
// fp32 [K,N] -> bf16 transposed [N,K], LDS-tiled 32x32 so both sides coalesce.
// block = (32,8); grid = (N/32, K/32)
// ---------------------------------------------------------------------------
__global__ __launch_bounds__(256) void transpose_cvt_bf16(const float* __restrict__ in,
                                                          __bf16* __restrict__ out,
                                                          int K, int N) {
    __shared__ float tile[32][33];
    const int tx = threadIdx.x, ty = threadIdx.y;
    const int n0 = blockIdx.x * 32;
    const int k0 = blockIdx.y * 32;
#pragma unroll
    for (int i = 0; i < 4; ++i)
        tile[ty + i * 8][tx] = in[(size_t)(k0 + ty + i * 8) * N + n0 + tx];
    __syncthreads();
#pragma unroll
    for (int i = 0; i < 4; ++i)
        out[(size_t)(n0 + ty + i * 8) * K + k0 + tx] = f2bf(tile[tx][ty + i * 8]);
}

// ---------------------------------------------------------------------------
// bf16 WMMA GEMM:  C[M,N] = A[M,K] @ Bt[N,K]^T + bias (+ optional ReLU)
//   A  : row-major [M,K] bf16
//   Bt : row-major [N,K] bf16 (pre-transposed weights)
// Block = 256 threads = 8 wave32 waves -> 128x64 block tile.
// Wave grid 4x2; each wave owns a 32x32 tile = 4 16x16 WMMA accumulators.
// Staging: double-buffered LDS fed by global_load_async_to_lds_b128,
// software-pipelined one k-step ahead; last iteration peeled so the hot
// loop is branch-free with an unconditional s_wait_asynccnt 3.
// ---------------------------------------------------------------------------
#define BM 128
#define BN 64
#define BK 32

template <bool OUT_BF16, bool RELU>
__global__ __launch_bounds__(256, 2) void gemm_wmma(const __bf16* __restrict__ A,
                                                    const __bf16* __restrict__ Bt,
                                                    const float* __restrict__ bias,
                                                    void* __restrict__ Cout,
                                                    int M, int N, int K) {
    __shared__ __bf16 As[2][BM * BK];   // [m][k]
    __shared__ __bf16 Bs[2][BN * BK];   // [n][k]

    const int tid  = threadIdx.x;
    const int wave = tid >> 5;
    const int lane = tid & 31;
    const int m0 = blockIdx.y * BM;
    const int n0 = blockIdx.x * BN;
    const int wm = (wave & 3) * 32;     // wave row offset inside block tile
    const int wn = (wave >> 2) * 32;    // wave col offset inside block tile

    // Async staging map: 16B segments along K (BK=32 bf16 = 64B = 4 segs/row).
    const int seg_r = tid >> 2;         // 0..63
    const int seg_c = (tid & 3) * 8;    // bf16 column offset

    // Running global pointers (advance by BK elements per k-step).
    const __bf16* ga0 = A + (size_t)(m0 + seg_r) * K + seg_c;
    const __bf16* ga1 = ga0 + (size_t)64 * K;
    const __bf16* gb  = Bt + (size_t)(n0 + seg_r) * K + seg_c;

    // Per-buffer LDS destinations (constant across the loop).
    unsigned la0[2], la1[2], lb[2];
#pragma unroll
    for (int u = 0; u < 2; ++u) {
        la0[u] = (unsigned)(uintptr_t)&As[u][seg_r * BK + seg_c];
        la1[u] = (unsigned)(uintptr_t)&As[u][(seg_r + 64) * BK + seg_c];
        lb[u]  = (unsigned)(uintptr_t)&Bs[u][seg_r * BK + seg_c];
    }

    v8f c00 = {}, c01 = {}, c10 = {}, c11 = {};

    // Fragment compute on buffer `cur` (ISA 16-bit A 16x32 / B 32x16 layouts).
    auto compute = [&](int cur) {
        // A: lane 0-15 -> M=lane,    K 0..7 (v0-3) & 16..23 (v4-7)
        //    lane16-31 -> M=lane-16, K 8..15        & 24..31
        const int am = wm + (lane & 15);
        const int kb = (lane >> 4) * 8;
        v16bf a0, a1;
        *reinterpret_cast<float4*>(&a0) =
            *reinterpret_cast<const float4*>(&As[cur][am * BK + kb]);
        *(reinterpret_cast<float4*>(&a0) + 1) =
            *reinterpret_cast<const float4*>(&As[cur][am * BK + kb + 16]);
        *reinterpret_cast<float4*>(&a1) =
            *reinterpret_cast<const float4*>(&As[cur][(am + 16) * BK + kb]);
        *(reinterpret_cast<float4*>(&a1) + 1) =
            *reinterpret_cast<const float4*>(&As[cur][(am + 16) * BK + kb + 16]);

        // B: lane 0-15 -> N=lane, K 0..7 & 16..23 ; lane 16-31 -> N=lane-16, K 8..15 & 24..31
        const int khalf = (lane >> 4) * 8;
        const int bn    = wn + (lane & 15);
        v16bf b0, b1;
        *reinterpret_cast<float4*>(&b0) =
            *reinterpret_cast<const float4*>(&Bs[cur][bn * BK + khalf]);
        *(reinterpret_cast<float4*>(&b0) + 1) =
            *reinterpret_cast<const float4*>(&Bs[cur][bn * BK + khalf + 16]);
        *reinterpret_cast<float4*>(&b1) =
            *reinterpret_cast<const float4*>(&Bs[cur][(bn + 16) * BK + khalf]);
        *(reinterpret_cast<float4*>(&b1) + 1) =
            *reinterpret_cast<const float4*>(&Bs[cur][(bn + 16) * BK + khalf + 16]);

        c00 = __builtin_amdgcn_wmma_f32_16x16x32_bf16(false, a0, false, b0,
                                                      (short)0, c00, false, false);
        c01 = __builtin_amdgcn_wmma_f32_16x16x32_bf16(false, a0, false, b1,
                                                      (short)0, c01, false, false);
        c10 = __builtin_amdgcn_wmma_f32_16x16x32_bf16(false, a1, false, b0,
                                                      (short)0, c10, false, false);
        c11 = __builtin_amdgcn_wmma_f32_16x16x32_bf16(false, a1, false, b1,
                                                      (short)0, c11, false, false);
    };

    // --- prologue: stage tile 0 into buffer 0 ---
    async_copy_b128(la0[0], ga0);
    async_copy_b128(la1[0], ga1);
    async_copy_b128(lb[0],  gb);

    const int nsteps = K / BK;
    // --- steady state: branch-free, always one tile in flight ---
    for (int t = 0; t < nsteps - 1; ++t) {
        const int cur = t & 1;
        const int nxt = cur ^ 1;
        ga0 += BK; ga1 += BK; gb += BK;

        __syncthreads();            // all waves done reading buf[nxt] (tile t-1)
        async_copy_b128(la0[nxt], ga0);
        async_copy_b128(la1[nxt], ga1);
        async_copy_b128(lb[nxt],  gb);
        WAIT_ASYNCCNT(3);           // tile t landed; tile t+1 in flight
        __syncthreads();            // tile t visible to every wave
        compute(cur);
    }
    // --- peeled final iteration ---
    __syncthreads();
    WAIT_ASYNCCNT(0);
    __syncthreads();
    compute((nsteps - 1) & 1);

    // Epilogue: C/D layout -- lane&15 = N column, VGPR j = row j (lo) / j+8 (hi)
    const int cn0  = n0 + wn + (lane & 15);
    const int mr0  = m0 + wm + ((lane >> 4) * 8);
    const float bs0 = bias[cn0];
    const float bs1 = bias[cn0 + 16];
#pragma unroll
    for (int j = 0; j < 8; ++j) {
        float x00 = c00[j] + bs0;
        float x01 = c01[j] + bs1;
        float x10 = c10[j] + bs0;
        float x11 = c11[j] + bs1;
        if (RELU) {
            x00 = fmaxf(x00, 0.f); x01 = fmaxf(x01, 0.f);
            x10 = fmaxf(x10, 0.f); x11 = fmaxf(x11, 0.f);
        }
        const size_t r0 = (size_t)(mr0 + j) * N;
        const size_t r1 = (size_t)(mr0 + 16 + j) * N;
        if (OUT_BF16) {
            __bf16* C = (__bf16*)Cout;
            C[r0 + cn0]      = f2bf(x00);
            C[r0 + cn0 + 16] = f2bf(x01);
            C[r1 + cn0]      = f2bf(x10);
            C[r1 + cn0 + 16] = f2bf(x11);
        } else {
            float* C = (float*)Cout;
            C[r0 + cn0]      = x00;
            C[r0 + cn0 + 16] = x01;
            C[r1 + cn0]      = x10;
            C[r1 + cn0 + 16] = x11;
        }
    }
}

// ---------------------------------------------------------------------------
// Sliding-window attention, window 65 (-32..+32), D=1024, fp32.
// One block per query. 8 waves split the offsets; wave32 shfl reductions.
// ---------------------------------------------------------------------------
#define ATT_S 2048
#define ATT_D 1024
#define ATT_HALF 32
#define ATT_W 65

__global__ __launch_bounds__(256, 2) void attn_kernel(const float* __restrict__ q,
                                                      const float* __restrict__ k,
                                                      const float* __restrict__ v,
                                                      float* __restrict__ out) {
    const int qidx = blockIdx.x;            // 0 .. B*S-1
    const int b = qidx / ATT_S;
    const int s = qidx % ATT_S;
    const int tid = threadIdx.x;
    const int lane = tid & 31;
    const int wave = tid >> 5;

    __shared__ float qs[ATT_D];
    __shared__ float sc[ATT_W];

    for (int d = tid; d < ATT_D; d += 256)
        qs[d] = q[(size_t)qidx * ATT_D + d];
    __syncthreads();

    const float scale = 0.03125f;  // 1/sqrt(1024)

    for (int j = wave; j < ATT_W; j += 8) {
        int idx = s + j - ATT_HALF;
        if (idx >= 0 && idx < ATT_S) {
            const float* kr = k + ((size_t)b * ATT_S + idx) * ATT_D;
            float p = 0.f;
            for (int d = lane; d < ATT_D; d += 32)
                p += qs[d] * kr[d];
#pragma unroll
            for (int o = 16; o > 0; o >>= 1)
                p += __shfl_xor(p, o, 32);
            if (lane == 0) sc[j] = p * scale;
        } else if (lane == 0) {
            sc[j] = -INFINITY;
        }
    }
    __syncthreads();

    if (tid == 0) {
        float mx = -INFINITY;
        for (int j = 0; j < ATT_W; ++j) mx = fmaxf(mx, sc[j]);
        float sum = 0.f;
        for (int j = 0; j < ATT_W; ++j) {
            float e = __expf(sc[j] - mx);
            sc[j] = e;
            sum += e;
        }
        float inv = 1.f / sum;
        for (int j = 0; j < ATT_W; ++j) sc[j] *= inv;
    }
    __syncthreads();

    for (int d = tid; d < ATT_D; d += 256) {
        float acc = 0.f;
        for (int j = 0; j < ATT_W; ++j) {
            int idx = s + j - ATT_HALF;
            if (idx >= 0 && idx < ATT_S)
                acc += sc[j] * v[((size_t)b * ATT_S + idx) * ATT_D + d];
        }
        out[(size_t)qidx * ATT_D + d] = acc;
    }
}

// ---------------------------------------------------------------------------
// Host-side launcher
// ---------------------------------------------------------------------------
extern "C" void kernel_launch(void* const* d_in, const int* in_sizes, int n_in,
                              void* d_out, int out_size, void* d_ws, size_t ws_size,
                              hipStream_t stream) {
    (void)in_sizes; (void)n_in; (void)out_size; (void)ws_size;

    constexpr int B = 2, S = 2048, D = 1024;
    constexpr int M = B * S;                 // 4096
    constexpr size_t MD = (size_t)M * D;     // 4M elements
    constexpr size_t DD = (size_t)D * D;     // 1M elements

    const float* x    = (const float*)d_in[0];
    const float* w1   = (const float*)d_in[1];
    const float* b1   = (const float*)d_in[2];
    const float* w2   = (const float*)d_in[3];
    const float* b2   = (const float*)d_in[4];
    const float* qw   = (const float*)d_in[5];
    const float* qb   = (const float*)d_in[6];
    const float* kw   = (const float*)d_in[7];
    const float* kb   = (const float*)d_in[8];
    const float* vw   = (const float*)d_in[9];
    const float* vb   = (const float*)d_in[10];
    float* outp = (float*)d_out;

    // Workspace carve-up (weights stored TRANSPOSED as [N,K] bf16)
    __bf16* xb  = (__bf16*)d_ws;             // M*D
    __bf16* w1t = xb  + MD;                  // D*D each
    __bf16* w2t = w1t + DD;
    __bf16* qwt = w2t + DD;
    __bf16* kwt = qwt + DD;
    __bf16* vwt = kwt + DD;
    __bf16* hb  = vwt + DD;                  // M*D
    __bf16* mb  = hb  + MD;                  // M*D
    float*  qf  = (float*)(mb + MD);         // M*D fp32
    float*  kf  = qf + MD;
    float*  vf  = kf + MD;

    // 1) conversions: x plain; weights transpose+convert
    {
        int n4x = (int)(MD / 4);
        cvt_f32_bf16_x4<<<(n4x + 255) / 256, 256, 0, stream>>>(x, xb, n4x);
        dim3 tgrid(D / 32, D / 32);
        dim3 tblock(32, 8);
        transpose_cvt_bf16<<<tgrid, tblock, 0, stream>>>(w1, w1t, D, D);
        transpose_cvt_bf16<<<tgrid, tblock, 0, stream>>>(w2, w2t, D, D);
        transpose_cvt_bf16<<<tgrid, tblock, 0, stream>>>(qw, qwt, D, D);
        transpose_cvt_bf16<<<tgrid, tblock, 0, stream>>>(kw, kwt, D, D);
        transpose_cvt_bf16<<<tgrid, tblock, 0, stream>>>(vw, vwt, D, D);
    }

    // 2) WMMA GEMM chain (async-LDS pipelined)
    dim3 ggrid(D / BN, M / BM);              // (16, 32)
    dim3 gblock(256);
    gemm_wmma<true,  true ><<<ggrid, gblock, 0, stream>>>(xb, w1t, b1, (void*)hb, M, D, D);
    gemm_wmma<true,  false><<<ggrid, gblock, 0, stream>>>(hb, w2t, b2, (void*)mb, M, D, D);
    gemm_wmma<false, false><<<ggrid, gblock, 0, stream>>>(mb, qwt, qb, (void*)qf, M, D, D);
    gemm_wmma<false, false><<<ggrid, gblock, 0, stream>>>(mb, kwt, kb, (void*)kf, M, D, D);
    gemm_wmma<false, false><<<ggrid, gblock, 0, stream>>>(mb, vwt, vb, (void*)vf, M, D, D);

    // 3) Sliding-window attention
    attn_kernel<<<M, 256, 0, stream>>>(qf, kf, vf, outp);
}
